// MaskedCRFLoss_60181081751607
// MI455X (gfx1250) — compile-verified, matrix operand order checked by reading, not run on hardware
//
#include <hip/hip_runtime.h>
#include <hip/hip_bf16.h>
#include <math.h>

typedef __attribute__((ext_vector_type(16))) _Float16 v16h;
typedef __attribute__((ext_vector_type(8)))  float    v8f;

#define S_LEN 512
#define B_SZ  256
#define T_SZ  128
#define BT    16          // batch tile per workgroup
#define FV_STRIDE 136     // pad: 136*4B row stride (16B aligned, bank-spread)
#define TRH_STRIDE 130    // pad: 130*2B = 260B row stride -> bank stride 65

// WMMA 16-bit fragment K index for element i of v16h at given lane.
// ISA 7.12.2: 16-bit A 16x32: VGPR r in 0..3: K = 2r+h (+8 for lanes 16..31);
// VGPR r in 4..7: K = 16 + 2(r-4)+h (+8 for lanes 16..31).  B (32x16) mirrors
// with lane -> N.  Element i -> VGPR i/2, half i%2.
__device__ __forceinline__ int frag_k(int i, int lane) {
  int r = i >> 1, h = i & 1;
  int base = ((r & 4) ? 16 : 0) + ((lane & 16) ? 8 : 0);
  return base + ((r & 3) << 1) + h;
}

__global__ __launch_bounds__(256) void crf_scan_kernel(
    const float* __restrict__ em, const float* __restrict__ mask,
    const float* __restrict__ trans, const float* __restrict__ start,
    const float* __restrict__ stop, float* __restrict__ out)
{
  __shared__ __align__(16) float    fv[BT][FV_STRIDE];     // forward var [b][j]
  __shared__ __align__(16) float    vvT[T_SZ][BT];         // viterbi var, transposed [j][b]
  __shared__ __align__(16) _Float16 ef[BT][FV_STRIDE];     // exp(fv - c)  [b][k]
  __shared__ __align__(16) _Float16 trH[T_SZ][TRH_STRIDE]; // transitions f16 [j][k]
  __shared__ float tmaxS[T_SZ];
  __shared__ float cS[BT];
  __shared__ float mS[BT];

  const int tid   = threadIdx.x;
  const int lane  = tid & 31;
  const int wave  = tid >> 5;        // 0..7 : owns j-tile [wave*16, wave*16+16)
  const int b0    = blockIdx.x * BT;
  const int jbase = wave * 16;
  const int n     = lane & 15;
  const int hi    = lane >> 4;       // 0 or 1
  const int jj    = jbase + n;       // this lane's output column

  // ---------------- init ----------------
  for (int idx = tid; idx < BT * T_SZ; idx += 256) {
    int b = idx >> 7, j = idx & 127;
    float v = start[j] + em[(size_t)(b0 + b) * T_SZ + j];
    fv[b][j]  = v;
    vvT[j][b] = v;
  }
  for (int idx = tid; idx < T_SZ * T_SZ; idx += 256)
    trH[idx >> 7][idx & 127] = (_Float16)trans[idx];
  if (tid < T_SZ) {
    float m = -INFINITY;
    for (int k = 0; k < T_SZ; k++) m = fmaxf(m, trans[tid * T_SZ + k]);
    tmaxS[tid] = m;
  }
  __syncthreads();

  // Per-wave constant B fragments: B[k][n] = exp(trans[jbase+n][k] - tmax) (f16),
  // held in registers for the entire scan (4 chunks of K=32 -> K=128).
  v16h Bf[4];
  for (int kc = 0; kc < 4; kc++) {
    #pragma unroll
    for (int i = 0; i < 16; i++) {
      int k = kc * 32 + frag_k(i, lane);
      Bf[kc][i] = (_Float16)__expf(trans[jj * T_SZ + k] - tmaxS[jj]);
    }
  }

  // Software-pipelined emission loads: this lane's 8 (m, jj) values for step t.
  float eCur[8];
  #pragma unroll
  for (int r = 0; r < 8; r++)
    eCur[r] = em[((size_t)1 * B_SZ + b0 + hi * 8 + r) * T_SZ + jj];

  // ---------------- sequential scan over time ----------------
  for (int t = 1; t < S_LEN; t++) {
    // phase 1: row maxes c[b] and ef[b][k] = exp(fv-c); stage this step's mask
    {
      int b = wave * 2;                       // each wave handles 2 batch rows
      #pragma unroll
      for (int bb = 0; bb < 2; bb++, b++) {
        int k0 = lane * 4;
        float4 x = *(const float4*)&fv[b][k0];
        float m = fmaxf(fmaxf(x.x, x.y), fmaxf(x.z, x.w));
        #pragma unroll
        for (int off = 16; off > 0; off >>= 1)
          m = fmaxf(m, __shfl_xor(m, off, 32));
        if (lane == 0) cS[b] = m;
        ef[b][k0 + 0] = (_Float16)__expf(x.x - m);
        ef[b][k0 + 1] = (_Float16)__expf(x.y - m);
        ef[b][k0 + 2] = (_Float16)__expf(x.z - m);
        ef[b][k0 + 3] = (_Float16)__expf(x.w - m);
      }
      if (wave == 0 && lane < BT) mS[lane] = mask[(size_t)t * B_SZ + b0 + lane];
    }
    __syncthreads();

    // phase 2a: WMMA  acc[b][j] = sum_k ef[b][k] * Et^T[k][j]
    v8f acc = {};
    #pragma unroll
    for (int kc = 0; kc < 4; kc++) {
      v16h a;
      #pragma unroll
      for (int i = 0; i < 16; i++)
        a[i] = ef[n][kc * 32 + frag_k(i, lane)];   // A: lane -> M(batch row)
      acc = __builtin_amdgcn_wmma_f32_16x16x32_f16(false, a, false, Bf[kc],
                                                   (short)0, acc, false, false);
    }

    // phase 2b: viterbi max-plus  vmx[m][j] = max_k (vv[m][k] + trans[j][k])
    float vmx[8];
    #pragma unroll
    for (int q = 0; q < 8; q++) vmx[q] = -INFINITY;
    for (int k = 0; k < T_SZ; k++) {
      float tr = (float)trH[jj][k];
      const float4* vp = (const float4*)&vvT[k][hi * 8];
      float4 x0 = vp[0], x1 = vp[1];
      vmx[0] = fmaxf(vmx[0], x0.x + tr);
      vmx[1] = fmaxf(vmx[1], x0.y + tr);
      vmx[2] = fmaxf(vmx[2], x0.z + tr);
      vmx[3] = fmaxf(vmx[3], x0.w + tr);
      vmx[4] = fmaxf(vmx[4], x1.x + tr);
      vmx[5] = fmaxf(vmx[5], x1.y + tr);
      vmx[6] = fmaxf(vmx[6], x1.z + tr);
      vmx[7] = fmaxf(vmx[7], x1.w + tr);
    }

    // prefetch pipeline: issue t+1 emission loads now (consumed next iteration),
    // and warm L2->WGP for t+2 (global_prefetch_b8).
    float eNext[8];
    if (t + 1 < S_LEN) {
      #pragma unroll
      for (int r = 0; r < 8; r++)
        eNext[r] = em[((size_t)(t + 1) * B_SZ + b0 + hi * 8 + r) * T_SZ + jj];
      if (t + 2 < S_LEN)
        __builtin_prefetch(&em[((size_t)(t + 2) * B_SZ + b0 + hi * 8) * T_SZ + jj], 0, 1);
    } else {
      #pragma unroll
      for (int r = 0; r < 8; r++) eNext[r] = 0.0f;
    }

    // phase 2c: emissions + mask blend (reads old fv/vv before overwrite)
    float fnew[8], vnew[8];
    #pragma unroll
    for (int r = 0; r < 8; r++) {              // C layout: m = r (+8 for hi half)
      int m = hi * 8 + r;
      float mk = mS[m];
      float e  = eCur[r];
      float nf = cS[m] + tmaxS[jj] + e + __logf(acc[r]);
      fnew[r] = mk * nf + (1.0f - mk) * fv[m][jj];
      float nv = vmx[r] + e;
      vnew[r] = mk * nv + (1.0f - mk) * vvT[jj][m];
    }
    __syncthreads();

    // phase 3: write back state
    #pragma unroll
    for (int r = 0; r < 8; r++) {
      fv[hi * 8 + r][jj]  = fnew[r];
      vvT[jj][hi * 8 + r] = vnew[r];
    }
    __syncthreads();

    #pragma unroll
    for (int r = 0; r < 8; r++) eCur[r] = eNext[r];
  }

  // ---------------- finalize: total = lse_j(fv+stop), best = max_j(vv+stop) ----------------
  {
    int b = wave * 2;
    for (int bb = 0; bb < 2; bb++, b++) {
      int j0 = lane * 4;
      float x[4], y[4];
      #pragma unroll
      for (int u = 0; u < 4; u++) {
        float sp = stop[j0 + u];
        x[u] = fv[b][j0 + u] + sp;
        y[u] = vvT[j0 + u][b] + sp;
      }
      float mmax = fmaxf(fmaxf(x[0], x[1]), fmaxf(x[2], x[3]));
      float bmax = fmaxf(fmaxf(y[0], y[1]), fmaxf(y[2], y[3]));
      #pragma unroll
      for (int off = 16; off > 0; off >>= 1) {
        mmax = fmaxf(mmax, __shfl_xor(mmax, off, 32));
        bmax = fmaxf(bmax, __shfl_xor(bmax, off, 32));
      }
      float s = __expf(x[0] - mmax) + __expf(x[1] - mmax) +
                __expf(x[2] - mmax) + __expf(x[3] - mmax);
      #pragma unroll
      for (int off = 16; off > 0; off >>= 1) s += __shfl_xor(s, off, 32);
      if (lane == 0) {
        out[1 + B_SZ     + b0 + b] = mmax + __logf(s); // total_score
        out[1 + 2 * B_SZ + b0 + b] = bmax;             // best_path_score
      }
    }
  }
}

// Gold (real) path score: one thread per batch row, serial gather over S.
__global__ __launch_bounds__(256) void crf_gold_kernel(
    const float* __restrict__ em, const float* __restrict__ mask,
    const int* __restrict__ tags, const float* __restrict__ trans,
    const float* __restrict__ start, const float* __restrict__ stop,
    float* __restrict__ out)
{
  int b = blockIdx.x * blockDim.x + threadIdx.x;
  if (b >= B_SZ) return;
  int prev = tags[b];
  float rps = start[prev] + em[(size_t)b * T_SZ + prev] * mask[b];
  for (int t = 1; t < S_LEN; t++) {
    int cur = tags[t * B_SZ + b];
    float m = mask[t * B_SZ + b];
    rps += (trans[prev * T_SZ + cur] +
            em[((size_t)t * B_SZ + b) * T_SZ + cur]) * m;
    prev = (m == 1.0f) ? cur : prev;
  }
  out[1 + b] = rps + stop[prev];
}

// loss = mean(total_score - real_path_score)
__global__ __launch_bounds__(256) void crf_loss_kernel(float* __restrict__ out)
{
  __shared__ float red[256];
  int tid = threadIdx.x;
  red[tid] = out[1 + B_SZ + tid] - out[1 + tid];
  __syncthreads();
  for (int s = 128; s > 0; s >>= 1) {
    if (tid < s) red[tid] += red[tid + s];
    __syncthreads();
  }
  if (tid == 0) out[0] = red[0] / (float)B_SZ;
}

extern "C" void kernel_launch(void* const* d_in, const int* in_sizes, int n_in,
                              void* d_out, int out_size, void* d_ws, size_t ws_size,
                              hipStream_t stream) {
  (void)in_sizes; (void)n_in; (void)d_ws; (void)ws_size; (void)out_size;
  const float* em    = (const float*)d_in[0];   // [S,B,T]
  const float* mask  = (const float*)d_in[1];   // [S,B]
  const int*   tags  = (const int*)d_in[2];     // [S,B]
  const float* trans = (const float*)d_in[3];   // [T,T]
  const float* start = (const float*)d_in[4];   // [T]
  const float* stop  = (const float*)d_in[5];   // [T]
  float* out = (float*)d_out;                   // [1 + 3*B]

  crf_scan_kernel<<<B_SZ / BT, 256, 0, stream>>>(em, mask, trans, start, stop, out);
  crf_gold_kernel<<<1, 256, 0, stream>>>(em, mask, tags, trans, start, stop, out);
  crf_loss_kernel<<<1, 256, 0, stream>>>(out);
}
